// WindowAttention_86474871538383
// MI455X (gfx1250) — matrix operand order repaired and checked by reading, not compiled, not run on hardware
//
#include <hip/hip_runtime.h>
#include <hip/hip_bf16.h>

// ---------------------------------------------------------------------------
// 3D window attention (Swin-style), fused flash-attention path for MI455X.
// All matmuls use v_wmma_f32_16x16x32_bf16 (bf16 inputs, fp32 accumulation).
// Requires ws_size >= ~139 MB.
// ---------------------------------------------------------------------------

typedef __attribute__((ext_vector_type(16))) __bf16 bf16x16;
typedef __attribute__((ext_vector_type(8)))  float  f32x8;

#define DIMC   192
#define HEADS  6
#define HD     32
#define NTOK   343
#define NPAD   352        // 11 * 32
#define BWIN   256        // B_ = BATCH * NUM_WIN
#define NWIN   64
#define MROWS  (BWIN * NTOK)      // 87808, divisible by 16

static __device__ __forceinline__ __bf16 f2bf(float f) { return (__bf16)f; }

// 16 contiguous bf16 (32B) as two b128 loads -- p must be 16B aligned.
static __device__ __forceinline__ bf16x16 load_b_frag(const __bf16* p) {
    union { bf16x16 v; uint4 u[2]; } t;
    t.u[0] = *reinterpret_cast<const uint4*>(p);
    t.u[1] = *reinterpret_cast<const uint4*>(p + 8);
    return t.v;
}
// A-fragment: elements 0..15 = row[base+sel*16 .. +15] (contiguous)
static __device__ __forceinline__ bf16x16 load_a_frag(const __bf16* row, int sel) {
    union { bf16x16 v; uint4 u[2]; } t;
    t.u[0] = *reinterpret_cast<const uint4*>(row + sel * 16);
    t.u[1] = *reinterpret_cast<const uint4*>(row + sel * 16 + 8);
    return t.v;
}

// ---------------------------------------------------------------------------
// Kernel 0: zero the padded V^T buffer (grid-stride).
// ---------------------------------------------------------------------------
__global__ void zero_bf16_kernel(__bf16* __restrict__ p, size_t n) {
    size_t i = (size_t)blockIdx.x * blockDim.x + threadIdx.x;
    size_t stride = (size_t)gridDim.x * blockDim.x;
    for (; i < n; i += stride) p[i] = f2bf(0.0f);
}

// ---------------------------------------------------------------------------
// Kernel 1: bias[h][q][k] = bias_table[rel_index[q*N+k]][h]
// ---------------------------------------------------------------------------
__global__ void bias_kernel(const float* __restrict__ table,
                            const int*  __restrict__ ridx,
                            float* __restrict__ biasH) {
    int idx = blockIdx.x * blockDim.x + threadIdx.x;
    if (idx < NTOK * NTOK) {
        int t = ridx[idx];
        #pragma unroll
        for (int h = 0; h < HEADS; ++h)
            biasH[(size_t)h * NTOK * NTOK + idx] = table[t * HEADS + h];
    }
}

// ---------------------------------------------------------------------------
// Kernel 2: QKV projection.  C[M=87808, 576] = x @ qkv_w^T + qkv_b.
// Per wave: 16x64 output tile, K-loop 192 in steps of 32 (6 x 4 WMMA).
// Scatters into head-separated bf16 buffers; Q pre-scaled by hd^-0.5,
// V stored transposed [b,h][d][n_pad] for contiguous PV B-fragment loads.
// ---------------------------------------------------------------------------
__global__ __launch_bounds__(128) void qkv_kernel(
    const float* __restrict__ x, const float* __restrict__ w,
    const float* __restrict__ bvec,
    __bf16* __restrict__ Q, __bf16* __restrict__ K, __bf16* __restrict__ Vt)
{
    const int lane = threadIdx.x & 31;
    const int wave = threadIdx.x >> 5;
    const int wid  = blockIdx.x * 4 + wave;        // 5488 * 9 waves total
    const int mt = wid / 9, nt = wid % 9;
    const int m0 = mt * 16, n0 = nt * 64;
    const int lm = lane & 15, sel = lane >> 4;

    f32x8 acc0 = {}, acc1 = {}, acc2 = {}, acc3 = {};
    f32x8* accp[4] = {&acc0, &acc1, &acc2, &acc3};

    const float* arow = x + (size_t)(m0 + lm) * DIMC;
    for (int kb = 0; kb < DIMC; kb += 32) {
        // A fragment (16x32, contiguous k = sel*16 + t)
        bf16x16 a;
        #pragma unroll
        for (int t = 0; t < 8; ++t) {
            a[t]     = f2bf(arow[kb + sel * 16 + t]);
            a[t + 8] = f2bf(arow[kb + sel * 16 + 8 + t]);
        }
        #pragma unroll
        for (int j = 0; j < 4; ++j) {
            const float* wr = w + (size_t)(n0 + 16 * j + lm) * DIMC + kb + sel * 16;
            bf16x16 bfr;
            #pragma unroll
            for (int t = 0; t < 16; ++t) bfr[t] = f2bf(wr[t]);
            *accp[j] = __builtin_amdgcn_wmma_f32_16x16x32_bf16(
                false, a, false, bfr, (short)0, *accp[j], false, false);
        }
    }

    const float qscale = 0.17677669529663687f;   // 32^-0.5
    #pragma unroll
    for (int j = 0; j < 4; ++j) {
        const int c    = n0 + 16 * j + lm;
        const int part = c / DIMC;               // 0=q 1=k 2=v
        const int cc   = c % DIMC;
        const int h    = cc / HD, d = cc % HD;
        const float bb = bvec[c];
        #pragma unroll
        for (int i = 0; i < 8; ++i) {
            const int r  = m0 + i + 8 * sel;     // C layout: row = i + 8*(lane>=16)
            const int bi = r / NTOK, nq = r % NTOK;
            const size_t bh = (size_t)(bi * HEADS + h);
            const float v = (*accp[j])[i] + bb;
            if (part == 0)      Q [(bh * NTOK + nq) * HD + d]   = f2bf(v * qscale);
            else if (part == 1) K [(bh * NTOK + nq) * HD + d]   = f2bf(v);
            else                Vt[(bh * HD + d) * NPAD + nq]   = f2bf(v);
        }
    }
}

// ---------------------------------------------------------------------------
// Kernel 3: fused attention (flash-style).  Block = (qtile, head, window).
// 4 waves x 16 query rows.  Per 32-key tile: 2 WMMA (S) + 2 WMMA (PV).
// ---------------------------------------------------------------------------
__global__ __launch_bounds__(128) void attn_kernel(
    const __bf16* __restrict__ Q,  const __bf16* __restrict__ Kb,
    const __bf16* __restrict__ Vt, const float* __restrict__ biasH,
    const float* __restrict__ mask, __bf16* __restrict__ Obuf)
{
    __shared__ alignas(16) __bf16 plds[4][16][32];   // per-wave P transpose buffer

    const int lane = threadIdx.x & 31, wave = threadIdx.x >> 5;
    const int lm = lane & 15, sel = lane >> 4;
    const int qt = blockIdx.x, h = blockIdx.y, b = blockIdx.z;
    const int q0 = qt * 64 + wave * 16;
    const size_t bh = (size_t)(b * HEADS + h);
    const __bf16* Qp = Q  + bh * NTOK * HD;
    const __bf16* Kp = Kb + bh * NTOK * HD;
    const __bf16* Vp = Vt + bh * HD * NPAD;
    const float*  bp = biasH + (size_t)h * NTOK * NTOK;
    const float*  mp = mask  + (size_t)(b & (NWIN - 1)) * NTOK * NTOK;

    // Q A-fragment (rows = lane%16), b128 loads
    bf16x16 qa;
    #pragma unroll
    for (int t = 0; t < 16; ++t) qa[t] = f2bf(0.0f);
    {
        const int q = q0 + lm;
        if (q < NTOK) qa = load_a_frag(Qp + (size_t)q * HD, sel);
    }

    float rowmax[8], rowsum[8];
    #pragma unroll
    for (int i = 0; i < 8; ++i) { rowmax[i] = -1e30f; rowsum[i] = 0.0f; }
    f32x8 o0 = {}, o1 = {};

    for (int kt = 0; kt < NPAD; kt += 32) {
        // --- S = Q @ K^T tile (16 x 32) ---
        f32x8 s0 = {}, s1 = {};
        #pragma unroll
        for (int j = 0; j < 2; ++j) {
            const int kr = kt + 16 * j + lm;
            bf16x16 bk;
            #pragma unroll
            for (int t = 0; t < 16; ++t) bk[t] = f2bf(0.0f);
            if (kr < NTOK) bk = load_b_frag(Kp + (size_t)kr * HD + sel * 16);
            f32x8 z = {};
            f32x8 sres = __builtin_amdgcn_wmma_f32_16x16x32_bf16(
                false, qa, false, bk, (short)0, z, false, false);
            if (j == 0) s0 = sres; else s1 = sres;
        }

        // --- bias + mask + online softmax ---
        #pragma unroll
        for (int i = 0; i < 8; ++i) {
            const int q = q0 + i + 8 * sel;
            const int k0 = kt + lm, k1 = kt + 16 + lm;
            float v0 = -1e30f, v1 = -1e30f;
            if (q < NTOK) {
                const size_t ro = (size_t)q * NTOK;
                if (k0 < NTOK) v0 = s0[i] + bp[ro + k0] + mp[ro + k0];
                if (k1 < NTOK) v1 = s1[i] + bp[ro + k1] + mp[ro + k1];
            }
            float t = fmaxf(v0, v1);
            t = fmaxf(t, __shfl_xor(t, 1));
            t = fmaxf(t, __shfl_xor(t, 2));
            t = fmaxf(t, __shfl_xor(t, 4));
            t = fmaxf(t, __shfl_xor(t, 8));
            const float nm   = fmaxf(rowmax[i], t);
            const float corr = __expf(rowmax[i] - nm);
            rowmax[i] = nm;
            const float p0 = __expf(v0 - nm), p1 = __expf(v1 - nm);
            s0[i] = p0; s1[i] = p1;
            float ts = p0 + p1;
            ts += __shfl_xor(ts, 1);
            ts += __shfl_xor(ts, 2);
            ts += __shfl_xor(ts, 4);
            ts += __shfl_xor(ts, 8);
            rowsum[i] = rowsum[i] * corr + ts;
            o0[i] *= corr; o1[i] *= corr;
        }

        // --- P (C layout) -> LDS -> A-fragment layout ---
        #pragma unroll
        for (int i = 0; i < 8; ++i) {
            const int m = i + 8 * sel;
            plds[wave][m][lm]      = f2bf(s0[i]);
            plds[wave][m][16 + lm] = f2bf(s1[i]);
        }
        __syncthreads();
        bf16x16 pa = load_a_frag(&plds[wave][lm][0], sel);   // two ds_load_b128
        __syncthreads();

        // --- O += P @ V (V^T rows are contiguous in k) ---
        #pragma unroll
        for (int j = 0; j < 2; ++j) {
            const int d = 16 * j + lm;
            bf16x16 bv = load_b_frag(Vp + (size_t)d * NPAD + kt + sel * 16);
            if (j == 0)
                o0 = __builtin_amdgcn_wmma_f32_16x16x32_bf16(
                        false, pa, false, bv, (short)0, o0, false, false);
            else
                o1 = __builtin_amdgcn_wmma_f32_16x16x32_bf16(
                        false, pa, false, bv, (short)0, o1, false, false);
        }
    }

    // --- normalize + store (bf16, [b][q][h*32+d] layout for proj GEMM) ---
    #pragma unroll
    for (int i = 0; i < 8; ++i) {
        const int q = q0 + i + 8 * sel;
        if (q < NTOK) {
            const float inv = 1.0f / rowsum[i];
            __bf16* orow = Obuf + ((size_t)(b * NTOK + q)) * DIMC + h * HD;
            orow[lm]      = f2bf(o0[i] * inv);
            orow[16 + lm] = f2bf(o1[i] * inv);
        }
    }
}

// ---------------------------------------------------------------------------
// Kernel 4: output projection.  out[M,192] = attn_out(bf16) @ proj_w^T + b.
// ---------------------------------------------------------------------------
__global__ __launch_bounds__(128) void proj_kernel(
    const __bf16* __restrict__ A, const float* __restrict__ w,
    const float* __restrict__ bvec, float* __restrict__ out)
{
    const int lane = threadIdx.x & 31;
    const int wave = threadIdx.x >> 5;
    const int wid  = blockIdx.x * 4 + wave;        // 5488 * 3 waves
    const int mt = wid / 3, nt = wid % 3;
    const int m0 = mt * 16, n0 = nt * 64;
    const int lm = lane & 15, sel = lane >> 4;

    f32x8 acc0 = {}, acc1 = {}, acc2 = {}, acc3 = {};
    f32x8* accp[4] = {&acc0, &acc1, &acc2, &acc3};

    const __bf16* arow = A + (size_t)(m0 + lm) * DIMC;
    for (int kb = 0; kb < DIMC; kb += 32) {
        bf16x16 a = load_a_frag(arow + kb, sel);           // b128 loads
        #pragma unroll
        for (int j = 0; j < 4; ++j) {
            const float* wr = w + (size_t)(n0 + 16 * j + lm) * DIMC + kb + sel * 16;
            bf16x16 bfr;
            #pragma unroll
            for (int t = 0; t < 16; ++t) bfr[t] = f2bf(wr[t]);
            *accp[j] = __builtin_amdgcn_wmma_f32_16x16x32_bf16(
                false, a, false, bfr, (short)0, *accp[j], false, false);
        }
    }
    #pragma unroll
    for (int j = 0; j < 4; ++j) {
        const int c = n0 + 16 * j + lm;
        const float bb = bvec[c];
        #pragma unroll
        for (int i = 0; i < 8; ++i) {
            const int r = m0 + i + 8 * sel;
            out[(size_t)r * DIMC + c] = (*accp[j])[i] + bb;
        }
    }
}

// ---------------------------------------------------------------------------
extern "C" void kernel_launch(void* const* d_in, const int* in_sizes, int n_in,
                              void* d_out, int out_size, void* d_ws, size_t ws_size,
                              hipStream_t stream) {
    const float* x          = (const float*)d_in[0];
    const float* mask       = (const float*)d_in[1];
    const float* qkv_w      = (const float*)d_in[2];
    const float* qkv_b      = (const float*)d_in[3];
    const float* proj_w     = (const float*)d_in[4];
    const float* proj_b     = (const float*)d_in[5];
    const float* bias_table = (const float*)d_in[6];
    const int*   rel_index  = (const int*)d_in[7];
    float* out = (float*)d_out;

    // workspace carve-up (256B aligned)
    const size_t QK_ELEMS   = (size_t)BWIN * HEADS * NTOK * HD;   // 16,859,136
    const size_t VT_ELEMS   = (size_t)BWIN * HEADS * HD * NPAD;   // 17,301,504
    const size_t BIAS_ELEMS = (size_t)HEADS * NTOK * NTOK;        //    705,894
    const size_t O_ELEMS    = (size_t)MROWS * DIMC;               // 16,859,136

    char* ws = (char*)d_ws;
    size_t off = 0;
    auto carve = [&](size_t bytes) -> char* {
        char* p = ws + off;
        off = (off + bytes + 255) & ~(size_t)255;
        return p;
    };
    __bf16* Qb    = (__bf16*)carve(QK_ELEMS * sizeof(__bf16));
    __bf16* Kbuf  = (__bf16*)carve(QK_ELEMS * sizeof(__bf16));
    __bf16* Vt    = (__bf16*)carve(VT_ELEMS * sizeof(__bf16));
    float*  biasH = (float*) carve(BIAS_ELEMS * sizeof(float));
    __bf16* Obuf  = (__bf16*)carve(O_ELEMS * sizeof(__bf16));
    (void)ws_size; (void)n_in; (void)in_sizes; (void)out_size;

    // 0) zero padded V^T
    zero_bf16_kernel<<<2048, 256, 0, stream>>>(Vt, VT_ELEMS);
    // 1) gather relative-position bias -> [H][N][N]
    bias_kernel<<<(NTOK * NTOK + 255) / 256, 256, 0, stream>>>(bias_table, rel_index, biasH);
    // 2) QKV projection (5488 Mtiles x 9 Ntiles, 4 waves/block)
    qkv_kernel<<<(MROWS / 16) * 9 / 4, 128, 0, stream>>>(x, qkv_w, qkv_b, Qb, Kbuf, Vt);
    // 3) fused flash attention: grid (qtiles=6, heads=6, windows=256)
    attn_kernel<<<dim3(6, HEADS, BWIN), 128, 0, stream>>>(Qb, Kbuf, Vt, biasH, mask, Obuf);
    // 4) output projection (5488 Mtiles x 3 Ntiles, 4 waves/block)
    proj_kernel<<<(MROWS / 16) * 3 / 4, 128, 0, stream>>>(Obuf, proj_w, proj_b, out);
}
